// Qwen3Attention_61615600828706
// MI455X (gfx1250) — matrix operand order repaired and checked by reading, not compiled, not run on hardware
//
#include <hip/hip_runtime.h>
#include <hip/hip_bf16.h>

typedef __attribute__((ext_vector_type(16))) _Float16 v16h;
typedef __attribute__((ext_vector_type(8)))  _Float16 v8h;
typedef __attribute__((ext_vector_type(4)))  _Float16 v4h;
typedef __attribute__((ext_vector_type(8)))  float    v8f;
typedef __attribute__((ext_vector_type(4)))  unsigned int u32x4;
typedef __attribute__((ext_vector_type(8)))  unsigned int u32x8;

#define LOG2E 1.4426950408889634f

__device__ __forceinline__ v16h cat8(v8h lo, v8h hi) {
    return __builtin_shufflevector(lo, hi, 0,1,2,3,4,5,6,7,8,9,10,11,12,13,14,15);
}

__device__ __forceinline__ v8f wmma_f16(v16h a, v16h b, v8f c) {
    return __builtin_amdgcn_wmma_f32_16x16x32_f16(
        false, a, false, b, (short)0, c, false, false);
}

// ---------------------------------------------------------------------------
// TDM: one-instruction DMA of a 2D f16 tile (tile_d0 contiguous elems x
// tile_d1 rows, row stride stride_d0 elems) from global into LDS.
// D# layout per CDNA5 ISA.  Tracked by TENSORcnt.
// ---------------------------------------------------------------------------
__device__ __forceinline__ void tdm_load_tile_f16(
    void* lds_dst, const void* gsrc,
    unsigned tile_d0, unsigned tile_d1, unsigned stride_d0)
{
    unsigned long long ga = (unsigned long long)(uintptr_t)gsrc;
    unsigned lds = (unsigned)(uintptr_t)lds_dst;   // flat low 32 = LDS offset
    u32x4 g0;
    g0[0] = 1u;                                        // count=1, no gather
    g0[1] = lds;                                       // lds_addr
    g0[2] = (unsigned)ga;                              // global_addr lo
    g0[3] = (unsigned)((ga >> 32) & 0x01FFFFFFull) | (2u << 30); // addr hi | type=2
    u32x8 g1;
    g1[0] = 1u << 16;                                  // data_size=2B, no flags
    g1[1] = (tile_d0 & 0xFFFFu) << 16;                 // tensor_dim0 lo16
    g1[2] = (tile_d0 >> 16) | ((tile_d1 & 0xFFFFu) << 16); // dim0 hi | dim1 lo
    g1[3] = (tile_d1 >> 16) | (tile_d0 << 16);         // dim1 hi | tile_dim0
    g1[4] = tile_d1 & 0xFFFFu;                         // tile_dim1, tile_dim2=0
    g1[5] = stride_d0;                                 // tensor_dim0_stride lo32
    g1[6] = 0u;                                        // stride hi | dim1_stride lo
    g1[7] = 0u;
    asm volatile("tensor_load_to_lds %0, %1" :: "s"(g0), "s"(g1) : "memory");
}

// ---------------------------------------------------------------------------
// GEMM: C[M,N] = A[M,K](f16) * Bt[N,K](f16, pre-transposed).  C f32.
// 256 threads = 8 waves, tile 128x128x64, double-buffered LDS fed by TDM.
// Each wave: 2 M-tiles x 4 N-tiles, 16 WMMAs per K-step.
// ---------------------------------------------------------------------------
__global__ __launch_bounds__(256)
void gemm_tdm_wmma_kernel(const _Float16* __restrict__ A,
                          const _Float16* __restrict__ Bt,
                          float* __restrict__ C, int M, int N, int K)
{
    (void)M;
    __shared__ _Float16 As[2][128][64];   // [m][k]
    __shared__ _Float16 Bs[2][128][64];   // [n][k]

    const int tid  = threadIdx.x;
    const int wid  = tid >> 5;
    const int lane = tid & 31;
    const int m    = lane & 15;
    const int h    = lane >> 4;
    const int wm   = wid & 3;
    const int wn   = wid >> 2;
    const int mBase = blockIdx.y * 128;
    const int nBase = blockIdx.x * 128;

    v8f acc[2][4];
    #pragma unroll
    for (int a = 0; a < 2; ++a)
        #pragma unroll
        for (int b = 0; b < 4; ++b)
            #pragma unroll
            for (int i = 0; i < 8; ++i) acc[a][b][i] = 0.0f;

    const int nk = K >> 6;  // K/64 steps
    if (wid == 0) {
        tdm_load_tile_f16(&As[0][0][0], A  + (size_t)mBase * K, 64, 128, (unsigned)K);
        tdm_load_tile_f16(&Bs[0][0][0], Bt + (size_t)nBase * K, 64, 128, (unsigned)K);
    }
    for (int i = 0; i < nk; ++i) {
        const int cur = i & 1;
        if (wid == 0) {
            if (i + 1 < nk) {
                tdm_load_tile_f16(&As[cur ^ 1][0][0],
                                  A  + (size_t)mBase * K + (i + 1) * 64, 64, 128, (unsigned)K);
                tdm_load_tile_f16(&Bs[cur ^ 1][0][0],
                                  Bt + (size_t)nBase * K + (i + 1) * 64, 64, 128, (unsigned)K);
                __builtin_amdgcn_s_wait_tensorcnt(2);  // tiles for step i done
            } else {
                __builtin_amdgcn_s_wait_tensorcnt(0);
            }
        }
        __syncthreads();
        #pragma unroll
        for (int c = 0; c < 2; ++c) {
            v16h af[2], bf[4];
            #pragma unroll
            for (int mt = 0; mt < 2; ++mt) {
                int row = wm * 32 + mt * 16 + m;
                af[mt] = cat8(*(const v8h*)&As[cur][row][32*c + 8*h],
                              *(const v8h*)&As[cur][row][32*c + 16 + 8*h]);
            }
            #pragma unroll
            for (int nt = 0; nt < 4; ++nt) {
                int col = wn * 64 + nt * 16 + m;
                bf[nt] = cat8(*(const v8h*)&Bs[cur][col][32*c + 8*h],
                              *(const v8h*)&Bs[cur][col][32*c + 16 + 8*h]);
            }
            #pragma unroll
            for (int mt = 0; mt < 2; ++mt)
                #pragma unroll
                for (int nt = 0; nt < 4; ++nt)
                    acc[mt][nt] = wmma_f16(af[mt], bf[nt], acc[mt][nt]);
        }
        __syncthreads();   // compute on 'cur' done before it is refilled
    }

    #pragma unroll
    for (int mt = 0; mt < 2; ++mt)
        #pragma unroll
        for (int nt = 0; nt < 4; ++nt)
            #pragma unroll
            for (int r = 0; r < 8; ++r)
                C[(size_t)(mBase + wm*32 + mt*16 + r + 8*h) * N
                  + nBase + wn*64 + nt*16 + m] = acc[mt][nt][r];
}

// ---------------------------------------------------------------------------
// f32 -> f16 convert (same layout)
// ---------------------------------------------------------------------------
__global__ __launch_bounds__(256)
void f32_to_f16_kernel(const float* __restrict__ in, _Float16* __restrict__ out, size_t n4)
{
    size_t i = (size_t)blockIdx.x * blockDim.x + threadIdx.x;
    if (i < n4) {
        float4 x = *(const float4*)(in + i * 4);
        v4h o;
        o[0] = (_Float16)x.x; o[1] = (_Float16)x.y;
        o[2] = (_Float16)x.z; o[3] = (_Float16)x.w;
        *(v4h*)(out + i * 4) = o;
    }
}

// ---------------------------------------------------------------------------
// W[K][N] f32 -> Wt[N][K] f16 via 64x64 LDS tile (coalesced both sides).
// Also used to pre-transpose V per batch: [S][1024] -> [1024][S].
// ---------------------------------------------------------------------------
__global__ __launch_bounds__(256)
void convert_transpose_kernel(const float* __restrict__ W, _Float16* __restrict__ Wt,
                              int K, int N)
{
    __shared__ _Float16 t[64][72];   // padded
    const int tid = threadIdx.x;
    const int k0 = blockIdx.y * 64, n0 = blockIdx.x * 64;
    #pragma unroll
    for (int i = 0; i < 4; ++i) {
        int lin = tid + 256 * i;        // 1024 float4 chunks
        int kr  = lin >> 4;
        int c4  = (lin & 15) << 2;
        float4 wv = *(const float4*)(W + (size_t)(k0 + kr) * N + n0 + c4);
        t[kr][c4+0] = (_Float16)wv.x;  t[kr][c4+1] = (_Float16)wv.y;
        t[kr][c4+2] = (_Float16)wv.z;  t[kr][c4+3] = (_Float16)wv.w;
    }
    __syncthreads();
    #pragma unroll
    for (int i = 0; i < 2; ++i) {
        int lin = tid + 256 * i;        // 512 chunks of 8 halves
        int nr  = lin >> 3;
        int kc  = (lin & 7) << 3;
        v8h o;
        #pragma unroll
        for (int j = 0; j < 8; ++j) o[j] = t[kc + j][nr];
        *(v8h*)(Wt + (size_t)(n0 + nr) * K + k0 + kc) = o;
    }
}

// ---------------------------------------------------------------------------
// Per-head RMSNorm + RoPE.  One wave per (token, head).
// ---------------------------------------------------------------------------
__global__ __launch_bounds__(256)
void rope_norm_kernel(const float* __restrict__ X, const float* __restrict__ cosb,
                      const float* __restrict__ sinb, const float* __restrict__ w,
                      _Float16* __restrict__ out, int nheads)
{
    const int tid  = threadIdx.x;
    const int wid  = tid >> 5;
    const int lane = tid & 31;
    const int token = blockIdx.x;
    const int head  = blockIdx.y * 8 + wid;

    const size_t base = ((size_t)token * nheads + head) * 128 + lane * 4;
    float4 x = *(const float4*)(X + base);

    float ss = x.x*x.x + x.y*x.y + x.z*x.z + x.w*x.w;
    ss += __shfl_xor(ss, 1, 32);
    ss += __shfl_xor(ss, 2, 32);
    ss += __shfl_xor(ss, 4, 32);
    ss += __shfl_xor(ss, 8, 32);
    ss += __shfl_xor(ss, 16, 32);
    const float r = rsqrtf(ss * (1.0f/128.0f) + 1e-6f);

    float4 wv = *(const float4*)(w + lane * 4);
    float xn0 = x.x * r * wv.x, xn1 = x.y * r * wv.y;
    float xn2 = x.z * r * wv.z, xn3 = x.w * r * wv.w;

    float p0 = __shfl_xor(xn0, 16, 32), p1 = __shfl_xor(xn1, 16, 32);
    float p2 = __shfl_xor(xn2, 16, 32), p3 = __shfl_xor(xn3, 16, 32);
    const float sgn = (lane < 16) ? -1.0f : 1.0f;

    float4 cv = *(const float4*)(cosb + (size_t)token * 128 + lane * 4);
    float4 sv = *(const float4*)(sinb + (size_t)token * 128 + lane * 4);

    v4h o;
    o[0] = (_Float16)(xn0 * cv.x + sgn * p0 * sv.x);
    o[1] = (_Float16)(xn1 * cv.y + sgn * p1 * sv.y);
    o[2] = (_Float16)(xn2 * cv.z + sgn * p2 * sv.z);
    o[3] = (_Float16)(xn3 * cv.w + sgn * p3 * sv.w);
    *(v4h*)(out + base) = o;
}

// ---------------------------------------------------------------------------
// Causal GQA flash attention.  128 threads = 4 waves; 64 queries per block.
// K blocks TDM-loaded [32 key][128 d]; V blocks TDM-loaded straight into the
// transposed fragment layout [128 d][32 key] from the pre-transposed global V.
// Double-buffered: block jb+1 DMA overlaps compute of block jb.
// ---------------------------------------------------------------------------
__global__ __launch_bounds__(128)
void attn_kernel(const _Float16* __restrict__ Q, const _Float16* __restrict__ Kb,
                 const _Float16* __restrict__ Vtg, _Float16* __restrict__ Ob)
{
    constexpr int S = 2048;
    __shared__ _Float16 Klds[2][32][128];   // [buf][key][d]
    __shared__ _Float16 Vt[2][128][32];     // [buf][d][key]
    __shared__ _Float16 Plds[4][16][32];

    const int tid  = threadIdx.x;
    const int wid  = tid >> 5;
    const int lane = tid & 31;
    const int m    = lane & 15;
    const int h    = lane >> 4;

    const int head = blockIdx.y;
    const int b    = blockIdx.z;
    const int kvh  = head >> 1;
    const int q0   = blockIdx.x * 64;
    const int qw   = q0 + wid * 16;

    // Vtg layout: [(b*8 + kvh)*128 + d][S]
    const size_t vbase = ((size_t)(b * 8 + kvh) * 128) * S;

    v16h qf[4];
    {
        const size_t qrow = (size_t)(b * S + qw + m) * 2048 + head * 128;
        #pragma unroll
        for (int c = 0; c < 4; ++c) {
            const _Float16* p = Q + qrow + 32 * c + 8 * h;
            qf[c] = cat8(*(const v8h*)p, *(const v8h*)(p + 16));
        }
    }

    float mi[8], li[8];
    v8f O[8];
    #pragma unroll
    for (int r = 0; r < 8; ++r) { mi[r] = -1e30f; li[r] = 0.0f; }
    #pragma unroll
    for (int t = 0; t < 8; ++t)
        #pragma unroll
        for (int r = 0; r < 8; ++r) O[t][r] = 0.0f;

    const int nblocks = blockIdx.x * 2 + 2;
    if (wid == 0) {
        tdm_load_tile_f16(&Klds[0][0][0], Kb + (size_t)(b * S) * 1024 + kvh * 128,
                          128, 32, 1024);
        tdm_load_tile_f16(&Vt[0][0][0],   Vtg + vbase, 32, 128, (unsigned)S);
    }
    for (int jb = 0; jb < nblocks; ++jb) {
        const int j0  = jb * 32;
        const int cur = jb & 1;
        if (wid == 0) {
            if (jb + 1 < nblocks) {
                const int jn = j0 + 32;
                tdm_load_tile_f16(&Klds[cur ^ 1][0][0],
                                  Kb + (size_t)(b * S + jn) * 1024 + kvh * 128,
                                  128, 32, 1024);
                tdm_load_tile_f16(&Vt[cur ^ 1][0][0],
                                  Vtg + vbase + jn, 32, 128, (unsigned)S);
                __builtin_amdgcn_s_wait_tensorcnt(2);
            } else {
                __builtin_amdgcn_s_wait_tensorcnt(0);
            }
        }
        __syncthreads();

        if (j0 <= qw + 15) {
            v8f s[2];
            #pragma unroll
            for (int t = 0; t < 2; ++t)
                #pragma unroll
                for (int r = 0; r < 8; ++r) s[t][r] = 0.0f;
            #pragma unroll
            for (int c = 0; c < 4; ++c) {
                #pragma unroll
                for (int t = 0; t < 2; ++t) {
                    const _Float16* p = &Klds[cur][t * 16 + m][32 * c + 8 * h];
                    v16h kf = cat8(*(const v8h*)p, *(const v8h*)(p + 16));
                    s[t] = wmma_f16(qf[c], kf, s[t]);
                }
            }
            const float sc = 0.08838834764831845f;  // 1/sqrt(128)
            #pragma unroll
            for (int t = 0; t < 2; ++t)
                #pragma unroll
                for (int r = 0; r < 8; ++r) {
                    int kcol = j0 + t * 16 + m;
                    int qrow = qw + r + 8 * h;
                    float v = s[t][r] * sc;
                    s[t][r] = (kcol <= qrow) ? v : -1e30f;
                }
            float rmax[8], rsum[8], alpha[8];
            #pragma unroll
            for (int r = 0; r < 8; ++r) rmax[r] = fmaxf(s[0][r], s[1][r]);
            #pragma unroll
            for (int msk = 1; msk < 16; msk <<= 1)
                #pragma unroll
                for (int r = 0; r < 8; ++r)
                    rmax[r] = fmaxf(rmax[r], __shfl_xor(rmax[r], msk, 32));
            #pragma unroll
            for (int r = 0; r < 8; ++r) {
                float mn = fmaxf(mi[r], rmax[r]);
                alpha[r] = exp2f((mi[r] - mn) * LOG2E);
                mi[r] = mn;
            }
            #pragma unroll
            for (int t = 0; t < 2; ++t)
                #pragma unroll
                for (int r = 0; r < 8; ++r)
                    s[t][r] = exp2f((s[t][r] - mi[r]) * LOG2E);
            #pragma unroll
            for (int r = 0; r < 8; ++r) rsum[r] = s[0][r] + s[1][r];
            #pragma unroll
            for (int msk = 1; msk < 16; msk <<= 1)
                #pragma unroll
                for (int r = 0; r < 8; ++r)
                    rsum[r] += __shfl_xor(rsum[r], msk, 32);
            #pragma unroll
            for (int r = 0; r < 8; ++r) li[r] = li[r] * alpha[r] + rsum[r];
            #pragma unroll
            for (int t = 0; t < 8; ++t)
                #pragma unroll
                for (int r = 0; r < 8; ++r) O[t][r] *= alpha[r];

            #pragma unroll
            for (int t = 0; t < 2; ++t)
                #pragma unroll
                for (int r = 0; r < 8; ++r)
                    Plds[wid][r + 8 * h][t * 16 + m] = (_Float16)s[t][r];
            const _Float16* pp = &Plds[wid][m][8 * h];
            v16h pf = cat8(*(const v8h*)pp, *(const v8h*)(&Plds[wid][m][16 + 8 * h]));

            #pragma unroll
            for (int t = 0; t < 8; ++t) {
                const _Float16* vp = &Vt[cur][t * 16 + m][8 * h];
                v16h vf = cat8(*(const v8h*)vp, *(const v8h*)(vp + 16));
                O[t] = wmma_f16(pf, vf, O[t]);
            }
        }
        __syncthreads();   // compute on 'cur' done before it is refilled
    }

    #pragma unroll
    for (int t = 0; t < 8; ++t)
        #pragma unroll
        for (int r = 0; r < 8; ++r) {
            float ov = O[t][r] / li[r];
            Ob[(size_t)(b * S + qw + r + 8 * h) * 2048 + head * 128 + t * 16 + m] =
                (_Float16)ov;
        }
}

// ---------------------------------------------------------------------------
extern "C" void kernel_launch(void* const* d_in, const int* in_sizes, int n_in,
                              void* d_out, int out_size, void* d_ws, size_t ws_size,
                              hipStream_t stream)
{
    (void)in_sizes; (void)n_in; (void)out_size; (void)ws_size;
    const float* hidden = (const float*)d_in[0];
    const float* cosb   = (const float*)d_in[1];
    const float* sinb   = (const float*)d_in[2];
    const float* Wq     = (const float*)d_in[3];
    const float* Wk     = (const float*)d_in[4];
    const float* Wv     = (const float*)d_in[5];
    const float* Wo     = (const float*)d_in[6];
    const float* qnw    = (const float*)d_in[7];
    const float* knw    = (const float*)d_in[8];
    float* out = (float*)d_out;

    char* ws = (char*)d_ws;
    const size_t MB = 1024u * 1024u;
    float*    qf32 = (float*)(ws);                  // 32 MB
    float*    kf32 = (float*)(ws + 32 * MB);        // 16 MB
    float*    vf32 = (float*)(ws + 48 * MB);        // 16 MB
    _Float16* qf16 = (_Float16*)(ws + 64 * MB);     // 16 MB
    _Float16* kf16 = (_Float16*)(ws + 80 * MB);     //  8 MB
    _Float16* Vtg  = (_Float16*)(ws + 88 * MB);     //  8 MB (pre-transposed V)
    _Float16* Xh   = (_Float16*)(ws + 96 * MB);     // 16 MB
    _Float16* Wqt  = (_Float16*)(ws + 112 * MB);    //  8 MB
    _Float16* Wkt  = (_Float16*)(ws + 120 * MB);    //  4 MB
    _Float16* Wvt  = (_Float16*)(ws + 124 * MB);    //  4 MB
    _Float16* Wot  = (_Float16*)(ws + 128 * MB);    //  8 MB
    _Float16* attn = (_Float16*)(ws);               // reuse dead qf32 region

    // precision prep: X -> f16, weights -> f16 transposed [N][K]
    f32_to_f16_kernel<<<dim3(8192), 256, 0, stream>>>(hidden, Xh, (size_t)4096 * 2048 / 4);
    convert_transpose_kernel<<<dim3(32, 32), 256, 0, stream>>>(Wq, Wqt, 2048, 2048);
    convert_transpose_kernel<<<dim3(16, 32), 256, 0, stream>>>(Wk, Wkt, 2048, 1024);
    convert_transpose_kernel<<<dim3(16, 32), 256, 0, stream>>>(Wv, Wvt, 2048, 1024);
    convert_transpose_kernel<<<dim3(32, 32), 256, 0, stream>>>(Wo, Wot, 2048, 2048);

    // QKV projections (TDM + WMMA)
    gemm_tdm_wmma_kernel<<<dim3(16, 32), 256, 0, stream>>>(Xh, Wqt, qf32, 4096, 2048, 2048);
    gemm_tdm_wmma_kernel<<<dim3( 8, 32), 256, 0, stream>>>(Xh, Wkt, kf32, 4096, 1024, 2048);
    gemm_tdm_wmma_kernel<<<dim3( 8, 32), 256, 0, stream>>>(Xh, Wvt, vf32, 4096, 1024, 2048);

    // RMSNorm + RoPE; V -> per-batch transposed f16 [1024][S]
    rope_norm_kernel<<<dim3(4096, 2), 256, 0, stream>>>(qf32, cosb, sinb, qnw, qf16, 16);
    rope_norm_kernel<<<dim3(4096, 1), 256, 0, stream>>>(kf32, cosb, sinb, knw, kf16, 8);
    convert_transpose_kernel<<<dim3(16, 32), 256, 0, stream>>>(vf32, Vtg, 2048, 1024);
    convert_transpose_kernel<<<dim3(16, 32), 256, 0, stream>>>(
        vf32 + (size_t)2048 * 1024, Vtg + (size_t)1024 * 2048, 2048, 1024);

    // causal GQA flash attention (double-buffered TDM K/V tiles + WMMA)
    attn_kernel<<<dim3(32, 16, 2), 128, 0, stream>>>(qf16, kf16, Vtg, attn);

    // output projection
    gemm_tdm_wmma_kernel<<<dim3(16, 32), 256, 0, stream>>>(attn, Wot, out, 4096, 2048, 2048);
}
